// VQVAE_36043365548313
// MI455X (gfx1250) — matrix operand order repaired and checked by reading, not compiled, not run on hardware
//
#include <hip/hip_runtime.h>
#include <stdint.h>

typedef _Float16 v16h __attribute__((ext_vector_type(16)));
typedef float    v8f  __attribute__((ext_vector_type(8)));

#define K_CODES 512
#define DIM      64
#define HW     1024          // 32*32 spatial per batch image
#define ROWS   128           // rows (N-vectors) per block
#define THREADS 256          // 8 waves of 32
#define XPAD    68           // padded row stride (halves) in LDS x tile

// ---------------------------------------------------------------------------
// Main kernel: distances via f16 WMMA, argmin, gather/output, loss partials.
// One block = 128 consecutive flat rows (same batch image, contiguous HW).
// ---------------------------------------------------------------------------
__global__ __launch_bounds__(THREADS)
void vq_main(const float* __restrict__ z, const float* __restrict__ codebook,
             float* __restrict__ out_q, float* __restrict__ out_idx,
             float* __restrict__ blockSums)
{
    // 32KB: codebook B-fragments (f16, dword-interleaved), 256 codes per half
    __shared__ uint32_t lds_b[8192];
    __shared__ _Float16 xtile[ROWS * XPAD];    // 17.4KB f16 x tile
    __shared__ float    enorm[K_CODES];        // fp32 ||e_k||^2
    __shared__ int      bestk[ROWS];
    __shared__ float    lss;

    const int tid = threadIdx.x;
    const int n0  = blockIdx.x * ROWS;         // first flat row
    const int b   = n0 >> 10;                  // batch image
    const int hw0 = n0 & 1023;                 // spatial offset
    const float* zb = z + (size_t)b * DIM * HW;

    // ---- stage x tile (f16) : coalesced along HW ----
    for (int it = 0; it < 32; ++it) {
        int pos = it * THREADS + tid;          // 0..8191
        int d = pos >> 7, r = pos & 127;
        xtile[r * XPAD + d] = (_Float16)zb[d * HW + hw0 + r];
    }
    // ---- fp32 code norms (codebook stays L2-resident) ----
    for (int kk = tid; kk < K_CODES; kk += THREADS) {
        const float4* row = (const float4*)(codebook + kk * DIM);
        float s = 0.f;
        #pragma unroll
        for (int i = 0; i < 16; ++i) {
            float4 v = row[i];
            s += v.x * v.x + v.y * v.y + v.z * v.z + v.w * v.w;
        }
        enorm[kk] = s;
    }
    if (tid == 0) lss = 0.f;
    __syncthreads();

    // ---- per-wave A fragments (16 rows x 64 K, contiguous k = 16*h + e order) ----
    const int lane = tid & 31;
    const int wv   = tid >> 5;                 // wave id -> M tile
    const int m    = lane & 15;
    const int h    = lane >> 4;
    const int row  = wv * 16 + m;

    v16h a0, a1;
    #pragma unroll
    for (int e = 0; e < 16; ++e) {
        int kk = e + 16 * h;                   // K index within 32
        a0[e] = xtile[row * XPAD + kk];        // K = 0..31
        a1[e] = xtile[row * XPAD + 32 + kk];   // K = 32..63
    }

    float bd[8];  int bi[8];
    #pragma unroll
    for (int i = 0; i < 8; ++i) { bd[i] = 3.4e38f; bi[i] = 0; }

    // ---- two halves of the codebook (256 codes each) through 32KB LDS ----
    for (int half = 0; half < 2; ++half) {
        // stage f16 B fragments, already in WMMA dword order:
        // dword pos = ((c*2+s)*8 + w)*32 + l  holds K=(s*32+16*(l/16)+2w, +1), N=c*16+(l%16)
        for (int it = 0; it < 32; ++it) {
            int pos = it * THREADS + tid;      // 0..8191
            int l = pos & 31, w = (pos >> 5) & 7, s = (pos >> 8) & 1, c = pos >> 9;
            int code = half * 256 + c * 16 + (l & 15);
            int d0   = s * 32 + (l >> 4) * 16 + 2 * w;
            float2 v = *(const float2*)(codebook + code * DIM + d0);
            union { _Float16 hh[2]; uint32_t u; } pk;
            pk.hh[0] = (_Float16)v.x; pk.hh[1] = (_Float16)v.y;
            lds_b[pos] = pk.u;
        }
        __syncthreads();

        #pragma unroll 2
        for (int c = 0; c < 16; ++c) {         // 16 code tiles of 16
            union { v16h hv; uint32_t u[8]; } b0, b1;
            #pragma unroll
            for (int w = 0; w < 8; ++w) {      // conflict-free ds_load_b32
                b0.u[w] = lds_b[((c * 2 + 0) * 8 + w) * 32 + lane];
                b1.u[w] = lds_b[((c * 2 + 1) * 8 + w) * 32 + lane];
            }
            // independent accumulators: no WMMA->WMMA RAW, both can co-execute
            v8f acc0 = {};
            v8f acc1 = {};
            acc0 = __builtin_amdgcn_wmma_f32_16x16x32_f16(
                       false, a0, false, b0.hv, (short)0, acc0, false, false);
            acc1 = __builtin_amdgcn_wmma_f32_16x16x32_f16(
                       false, a1, false, b1.hv, (short)0, acc1, false, false);

            int   cand = half * 256 + c * 16 + m;   // N = lane%16 in C layout
            float en   = enorm[cand];
            #pragma unroll
            for (int i = 0; i < 8; ++i) {
                float dot  = acc0[i] + acc1[i];
                float dist = fmaf(-2.0f, dot, en);    // ||e||^2 - 2 x.e
                bool  take = dist < bd[i];
                bd[i] = take ? dist : bd[i];
                bi[i] = take ? cand : bi[i];
            }
        }
        __syncthreads();                        // WAR on lds_b before restage
    }

    // ---- argmin across the 16 lanes sharing each M row (half-wave shuffle) ----
    #pragma unroll
    for (int i = 0; i < 8; ++i) {
        #pragma unroll
        for (int off = 1; off < 16; off <<= 1) {
            float od = __shfl_xor(bd[i], off, 16);
            int   oi = __shfl_xor(bi[i], off, 16);
            // branchless min-with-index, first-index tie-break (matches argmin)
            bool take = (od < bd[i]) | ((od == bd[i]) & (oi < bi[i]));
            bd[i] = take ? od : bd[i];
            bi[i] = take ? oi : bi[i];
        }
    }
    if (m == 0) {                               // lane 0 -> rows i, lane 16 -> rows i+8
        #pragma unroll
        for (int i = 0; i < 8; ++i) bestk[wv * 16 + h * 8 + i] = bi[i];
    }
    __syncthreads();

    // ---- output phase: coalesced NCHW store, fp32 loss from fresh global x ----
    float lsum = 0.f;
    for (int it = 0; it < 32; ++it) {
        int pos = it * THREADS + tid;
        int d = pos >> 7, r = pos & 127;
        int k = bestk[r];
        float q  = codebook[k * DIM + d];       // L2-resident gather
        float xv = zb[d * HW + hw0 + r];
        float df = q - xv;
        lsum += df * df;
        out_q[(size_t)b * DIM * HW + d * HW + hw0 + r] = q;
    }
    if (tid < ROWS) out_idx[n0 + tid] = (float)bestk[tid];

    atomicAdd(&lss, lsum);
    __syncthreads();
    if (tid == 0) blockSums[blockIdx.x] = lss;
}

// ---------------------------------------------------------------------------
// Reduce 512 per-block loss partials -> loss = 1.25 * mean((q - x)^2)
// ---------------------------------------------------------------------------
__global__ __launch_bounds__(256)
void vq_loss_finalize(const float* __restrict__ partial, float* __restrict__ loss_out)
{
    __shared__ float sdata[256];
    float s = partial[threadIdx.x] + partial[threadIdx.x + 256];
    sdata[threadIdx.x] = s;
    __syncthreads();
    for (int stride = 128; stride > 0; stride >>= 1) {
        if (threadIdx.x < stride) sdata[threadIdx.x] += sdata[threadIdx.x + stride];
        __syncthreads();
    }
    if (threadIdx.x == 0)
        loss_out[0] = 1.25f * sdata[0] / 4194304.0f;   // mean over N*D = 65536*64
}

extern "C" void kernel_launch(void* const* d_in, const int* in_sizes, int n_in,
                              void* d_out, int out_size, void* d_ws, size_t ws_size,
                              hipStream_t stream)
{
    (void)in_sizes; (void)n_in; (void)out_size; (void)ws_size;
    const float* z        = (const float*)d_in[0];   // [64,64,32,32] fp32
    const float* codebook = (const float*)d_in[1];   // [512,64] fp32

    float* out   = (float*)d_out;                    // quantized NCHW: 4194304 floats
    float* loss  = out + 4194304;                    // 1 float
    float* idxf  = loss + 1;                         // 65536 floats (indices)
    float* bsums = (float*)d_ws;                     // 512 partials

    vq_main<<<512, THREADS, 0, stream>>>(z, codebook, out, idxf, bsums);
    vq_loss_finalize<<<1, 256, 0, stream>>>(bsums, loss);
}
